// ExtSummModel_78477642433187
// MI455X (gfx1250) — matrix operand order repaired and checked by reading, not compiled, not run on hardware
//
#include <hip/hip_runtime.h>
#include <hip/hip_bf16.h>

// ---------------------------------------------------------------------------
// Model dims (compile-time constants from the reference)
// ---------------------------------------------------------------------------
#define BB   32
#define SS   512
#define LL   20
#define TT   16
#define EE   300
#define EP   320        // E padded to multiple of 32 for WMMA K
#define HH   128
#define H3   384        // 3*H
#define H4   512        // 4*H
#define DD   128
#define ROWS (BB*SS)    // 16384

typedef __attribute__((ext_vector_type(16))) _Float16 v16h;
typedef __attribute__((ext_vector_type(8)))  _Float16 v8h;
typedef __attribute__((ext_vector_type(8)))  float    v8f;
typedef __attribute__((ext_vector_type(4)))  unsigned int u32x4;
typedef __attribute__((ext_vector_type(8)))  int          i32x8;
typedef __attribute__((ext_vector_type(4)))  int          i32x4;

#if defined(__has_builtin)
#if __has_builtin(__builtin_amdgcn_tensor_load_to_lds) && \
    __has_builtin(__builtin_amdgcn_s_wait_tensorcnt)
#define HAVE_TDM 1
#endif
#endif

// ---------------------------------------------------------------------------
// WMMA fragment loaders (16-bit A/B layouts per CDNA5 ISA 7.12.2)
// A: row-major (M x K); lane<16 : row m0+lane, K k0+{0..7} & k0+16+{0..7}
//                       lane>=16: row m0+lane-16, K k0+8+{0..7} & k0+24+{0..7}
// B: fed from BT = (N x K) row-major; same K split as A: lane<16 holds K
//    k0+{0..7} & k0+16+{0..7} of col n0+lane; lane>=16 k0+8+{0..7} & k0+24+{0..7}.
// ---------------------------------------------------------------------------
__device__ __forceinline__ v16h load_a_frag(const _Float16* A, int lda,
                                            int m0, int k0, int r, int half) {
  const _Float16* p = A + (size_t)(m0 + r) * lda + k0 + half * 8;
  v16h a;
  ((v8h*)&a)[0] = *(const v8h*)p;
  ((v8h*)&a)[1] = *(const v8h*)(p + 16);
  return a;
}

__device__ __forceinline__ v16h load_b_frag(const _Float16* BT, int ldk,
                                            int n0, int k0, int r, int half) {
  const _Float16* p = BT + (size_t)(n0 + r) * ldk + k0 + half * 8;
  v16h b;
  ((v8h*)&b)[0] = *(const v8h*)p;
  ((v8h*)&b)[1] = *(const v8h*)(p + 16);
  return b;
}

#define WMMA(a, b, c) \
  __builtin_amdgcn_wmma_f32_16x16x32_f16(false, (a), false, (b), (short)0, (c), false, false)

// ---------------------------------------------------------------------------
// Generic WMMA GEMM: C(MxN,f32) = A(MxK,f16 rm) * BT(NxK,f16)^T + bias
// One 16(M) x 64(N) strip per wave: 1 A-frag + 4 B-frags -> 4 WMMAs per
// K-step; K-loop software-pipelined (next frags in flight over current math).
// N must be a multiple of 64.
// ---------------------------------------------------------------------------
__global__ void wmma_gemm_f16(const _Float16* __restrict__ A,
                              const _Float16* __restrict__ BT,
                              const float* __restrict__ bias,
                              float* __restrict__ C,
                              int M, int N, int K, int lda, int ldk, int ldc) {
  const int wid  = (blockIdx.x * blockDim.x + threadIdx.x) >> 5;
  const int lane = threadIdx.x & 31;
  const int strips_n = N >> 6;
  const int tm = (wid / strips_n) << 4;
  const int tn = (wid % strips_n) << 6;
  if (tm >= M) return;
  const int r = lane & 15, half = lane >> 4;

  v8f c0 = {}, c1 = {}, c2 = {}, c3 = {};

  v16h a  = load_a_frag(A, lda, tm, 0, r, half);
  v16h b0 = load_b_frag(BT, ldk, tn,      0, r, half);
  v16h b1 = load_b_frag(BT, ldk, tn + 16, 0, r, half);
  v16h b2 = load_b_frag(BT, ldk, tn + 32, 0, r, half);
  v16h b3 = load_b_frag(BT, ldk, tn + 48, 0, r, half);

  int k0 = 0;
  while (k0 + 32 < K) {
    const int kn = k0 + 32;
    if (kn + 32 < K) {
      __builtin_prefetch(A  + (size_t)(tm + r) * lda + kn + 32, 0, 1);
      __builtin_prefetch(BT + (size_t)(tn + r) * ldk + kn + 32, 0, 1);
    }
    v16h an  = load_a_frag(A, lda, tm, kn, r, half);
    v16h bn0 = load_b_frag(BT, ldk, tn,      kn, r, half);
    v16h bn1 = load_b_frag(BT, ldk, tn + 16, kn, r, half);
    v16h bn2 = load_b_frag(BT, ldk, tn + 32, kn, r, half);
    v16h bn3 = load_b_frag(BT, ldk, tn + 48, kn, r, half);
    c0 = WMMA(a, b0, c0);
    c1 = WMMA(a, b1, c1);
    c2 = WMMA(a, b2, c2);
    c3 = WMMA(a, b3, c3);
    a = an; b0 = bn0; b1 = bn1; b2 = bn2; b3 = bn3;
    k0 = kn;
  }
  c0 = WMMA(a, b0, c0);
  c1 = WMMA(a, b1, c1);
  c2 = WMMA(a, b2, c2);
  c3 = WMMA(a, b3, c3);

  v8f acc[4] = {c0, c1, c2, c3};
#pragma unroll
  for (int s = 0; s < 4; ++s) {
    const int n = tn + s * 16 + r;
    const float bv = bias ? bias[n] : 0.0f;
#pragma unroll
    for (int j = 0; j < 8; ++j)
      C[(size_t)(tm + j + half * 8) * ldc + n] = acc[s][j] + bv;
  }
}

// ---------------------------------------------------------------------------
// Sentence encoder: mean of word embeddings, write f16 padded (ROWS x EP)
// ---------------------------------------------------------------------------
__global__ void embed_mean_kernel(const int* __restrict__ word_ids,
                                  const float* __restrict__ table,
                                  _Float16* __restrict__ sent16) {
  const int row = blockIdx.x;
  const int* w = word_ids + (size_t)row * LL;
  for (int e = threadIdx.x; e < EP; e += blockDim.x) {
    float acc = 0.0f;
    if (e < EE) {
#pragma unroll
      for (int l = 0; l < LL; ++l)
        acc += table[(size_t)w[l] * EE + e];
      acc *= (1.0f / (float)LL);
    }
    sent16[(size_t)row * EP + e] = (_Float16)acc;
  }
}

// f32 (rows x cols) -> f16 (rows x colsPad), zero-padded
__global__ void convert_pad_f16(const float* __restrict__ src,
                                _Float16* __restrict__ dst,
                                int cols, int colsPad) {
  const int r = blockIdx.x;
  for (int c = threadIdx.x; c < colsPad; c += blockDim.x)
    dst[(size_t)r * colsPad + c] =
        (_Float16)(c < cols ? src[(size_t)r * cols + c] : 0.0f);
}

// f32 (K x N) -> f16 transposed (N x K)
__global__ void transpose_to_f16(const float* __restrict__ src,
                                 _Float16* __restrict__ dst, int K, int N) {
  const int n = blockIdx.x;
  for (int k = threadIdx.x; k < K; k += blockDim.x)
    dst[(size_t)n * K + k] = (_Float16)src[(size_t)k * N + n];
}

// ---------------------------------------------------------------------------
// Persistent bidirectional GRU: grid of 2 (dir = blockIdx.x), 1024 threads.
// LDS: Wh f16 (384x128, 96KB, staged once via the Tensor Data Mover)
//    + h f32 (32x128, 16KB) + gh f32 (32x384, 48KB)  => 160KB dynamic.
// Per step: gh = h @ Wh^T via WMMA (A converted f32->f16 from LDS on the fly,
// B fragments are ds_load_b128 from the staged LDS panel), then elementwise
// torch-order gates (r,z,n).
// ---------------------------------------------------------------------------
#define GRU_LDS_BYTES (H3*HH*2 + BB*HH*4 + BB*H3*4)

__global__ void gru_kernel(const float* __restrict__ GIf, const float* __restrict__ GIb,
                           const _Float16* __restrict__ Wh16f, const _Float16* __restrict__ Wh16b,
                           const float* __restrict__ bhf, const float* __restrict__ bhb,
                           float* __restrict__ outf, float* __restrict__ outb) {
  extern __shared__ char smraw[];
  _Float16* whlds = (_Float16*)smraw;                       // 384*128 f16
  float*    hbuf  = (float*)(smraw + H3 * HH * 2);          // 32*128  f32
  float*    gh    = (float*)(smraw + H3 * HH * 2 + BB * HH * 4); // 32*384 f32

  const int dir = blockIdx.x;
  const float*    GI  = dir ? GIb   : GIf;
  const _Float16* Wh  = dir ? Wh16b : Wh16f;
  const float*    bh  = dir ? bhb   : bhf;
  float*          out = dir ? outb  : outf;

  const int tid = threadIdx.x;
  const int wid = tid >> 5, lane = tid & 31;

  // ---- stage Wh panel into LDS (TDM if available, else cooperative copy) ---
#if defined(HAVE_TDM)
  if (wid == 0) {
    const unsigned long long ga = (unsigned long long)(uintptr_t)Wh;
    const unsigned int ldsaddr  = (unsigned int)(uintptr_t)whlds;
    u32x4 g0;
    g0.x = 1u;                                       // count=1, user mode
    g0.y = ldsaddr;                                  // lds_addr
    g0.z = (unsigned int)(ga & 0xFFFFFFFFu);         // global_addr[31:0]
    g0.w = (unsigned int)((ga >> 32) & 0x1FFFFFFu)   // global_addr[56:32]
         | (2u << 30);                               // type=2 ("image")
    i32x8 g1;
    g1[0] = 0x10000;          // data_size=1 (2 bytes/elem)
    g1[1] = (HH << 16);       // tensor_dim0[15:0] = 128 (bits 63:48)
    g1[2] = (H3 << 16);       // tensor_dim1[15:0] = 384 (bits 95:80)
    g1[3] = (HH << 16);       // tile_dim0 = 128 (bits 127:112)
    g1[4] = H3;               // tile_dim1 = 384 (bits 143:128)
    g1[5] = HH;               // tensor_dim0_stride[31:0] = 128
    g1[6] = 0;
    g1[7] = 0;
    const i32x4 z4 = {0, 0, 0, 0};
#if __clang_major__ >= 23
    const i32x8 z8 = {0, 0, 0, 0, 0, 0, 0, 0};
    __builtin_amdgcn_tensor_load_to_lds(g0, g1, z4, z4, z8, 0);
#else
    __builtin_amdgcn_tensor_load_to_lds(g0, g1, z4, z4, 0);
#endif
    __builtin_amdgcn_s_wait_tensorcnt(0);
  }
#else
  for (int i = tid; i < H3 * HH; i += 1024) whlds[i] = Wh[i];
#endif

  for (int i = tid; i < BB * HH; i += 1024) hbuf[i] = 0.0f;
  __syncthreads();

  const int r = lane & 15, half = lane >> 4;

  for (int step = 0; step < SS; ++step) {
    const int t = dir ? (SS - 1 - step) : step;

    // --- gh = h @ Wh^T : M=32, N=384, K=128 -> 48 tiles over 32 waves ---
    for (int tile = wid; tile < 48; tile += 32) {
      const int tm = (tile / 24) << 4;
      const int tn = (tile % 24) << 4;
      const float* ph = hbuf + (size_t)(tm + r) * HH;
      v8f c = {};
      for (int k0 = 0; k0 < HH; k0 += 32) {
        v16h a;
        _Float16* ap = (_Float16*)&a;
#pragma unroll
        for (int i = 0; i < 8; ++i) {
          ap[i]     = (_Float16)ph[k0 + half * 8 + i];
          ap[8 + i] = (_Float16)ph[k0 + 16 + half * 8 + i];
        }
        v16h b = load_b_frag(whlds, HH, tn, k0, r, half);
        c = WMMA(a, b, c);
      }
#pragma unroll
      for (int j = 0; j < 8; ++j)
        gh[(tm + j + half * 8) * H3 + tn + r] = c[j];
    }
    __syncthreads();

    // --- elementwise gates (torch order r,z,n) ---
    for (int idx = tid; idx < BB * HH; idx += 1024) {
      const int b  = idx >> 7;
      const int hc = idx & 127;
      const float* gi = GI + ((size_t)b * SS + t) * H3;
      const float gr = gh[b * H3 + hc]          + bh[hc];
      const float gz = gh[b * H3 + HH + hc]     + bh[HH + hc];
      const float gn = gh[b * H3 + 2 * HH + hc] + bh[2 * HH + hc];
      const float rr = 1.0f / (1.0f + __expf(-(gi[hc] + gr)));
      const float zz = 1.0f / (1.0f + __expf(-(gi[HH + hc] + gz)));
      const float nn = tanhf(gi[2 * HH + hc] + rr * gn);
      const float hn = (1.0f - zz) * nn + zz * hbuf[idx];
      hbuf[idx] = hn;
      out[((size_t)b * SS + t) * HH + hc] = hn;
    }
    __syncthreads();
  }
}

// ---------------------------------------------------------------------------
// Topic boundary diffs: topic_mat (B,T,2H). 1-based padded indexing guards.
// ---------------------------------------------------------------------------
__global__ void topic_kernel(const int* __restrict__ starts, const int* __restrict__ ends,
                             const float* __restrict__ outf, const float* __restrict__ outb,
                             float* __restrict__ topic_mat) {
  const int bt = blockIdx.x;
  const int b  = bt >> 4;  // T=16
  const int st = starts[bt], en = ends[bt];
  for (int c = threadIdx.x; c < 2 * HH; c += blockDim.x) {
    float val;
    if (c < HH) {
      const float a = (en >= 1 && en <= SS) ? outf[((size_t)b * SS + en - 1) * HH + c] : 0.0f;
      const int   p = st - 1;
      const float d = (p >= 1 && p <= SS) ? outf[((size_t)b * SS + p - 1) * HH + c] : 0.0f;
      val = a - d;
    } else {
      const int cc = c - HH;
      const float a = (st >= 1 && st <= SS) ? outb[((size_t)b * SS + st - 1) * HH + cc] : 0.0f;
      const int   p = en + 1;
      const float d = (p >= 1 && p <= SS) ? outb[((size_t)b * SS + p - 1) * HH + cc] : 0.0f;
      val = a - d;
    }
    topic_mat[(size_t)bt * (2 * HH) + c] = val;
  }
}

// doc_rep faithful to torch (2,B,H).view(B,2H) quirk
__device__ __forceinline__ float doc_rep_at(const float* outf, const float* outb,
                                            int b, int c) {
  const int f = b * 256 + c;
  if (f < BB * HH)
    return outf[((size_t)(f >> 7) * SS + (SS - 1)) * HH + (f & 127)];
  const int g = f - BB * HH;
  return outb[((size_t)(g >> 7) * SS) * HH + (g & 127)];
}

// Build cat_ds / cat_ts rows in f16 for the attention GEMMs
__global__ void build_cat_kernel(const float* __restrict__ outf, const float* __restrict__ outb,
                                 const float* __restrict__ topic_mat,
                                 const int* __restrict__ sent2topic,
                                 _Float16* __restrict__ catds, _Float16* __restrict__ catts) {
  const int row = blockIdx.x;
  const int b = row >> 9;
  const int tp = sent2topic[row];
  for (int c = threadIdx.x; c < H4; c += blockDim.x) {
    float vds, vts;
    if (c < 2 * HH) {
      vds = doc_rep_at(outf, outb, b, c);
      vts = topic_mat[((size_t)b * TT + tp) * (2 * HH) + c];
    } else {
      const int cc = c - 2 * HH;
      const float sr = (cc < HH) ? outf[(size_t)row * HH + cc]
                                 : outb[(size_t)row * HH + cc - HH];
      vds = sr; vts = sr;
    }
    catds[(size_t)row * H4 + c] = (_Float16)vds;
    catts[(size_t)row * H4 + c] = (_Float16)vts;
  }
}

// Scores (tanh(Y) . v_att), normalized weights, context, build inp f16 rows
__global__ void score_context_kernel(const float* __restrict__ Yds, const float* __restrict__ Yts,
                                     const float* __restrict__ vatt,
                                     const float* __restrict__ outf, const float* __restrict__ outb,
                                     const float* __restrict__ topic_mat,
                                     const int* __restrict__ sent2topic,
                                     _Float16* __restrict__ inp16) {
  const int row  = blockIdx.x * 8 + (threadIdx.x >> 5);
  const int lane = threadIdx.x & 31;
  float ds = 0.0f, ts = 0.0f;
  for (int i = lane; i < H4; i += 32) {
    const float v = vatt[i];
    ds += tanhf(Yds[(size_t)row * H4 + i]) * v;
    ts += tanhf(Yts[(size_t)row * H4 + i]) * v;
  }
  for (int off = 16; off > 0; off >>= 1) {
    ds += __shfl_xor(ds, off, 32);
    ts += __shfl_xor(ts, off, 32);
  }
  const float sum = ds + ts;
  const float dw = ds / sum, tw = ts / sum;
  const int b  = row >> 9;
  const int tp = sent2topic[row];
  for (int c = lane; c < 2 * HH; c += 32) {
    const float dr  = doc_rep_at(outf, outb, b, c);
    const float tr  = topic_mat[((size_t)b * TT + tp) * (2 * HH) + c];
    const float ctx = dw * dr + tw * tr;
    const float sr  = (c < HH) ? outf[(size_t)row * HH + c]
                               : outb[(size_t)row * HH + c - HH];
    inp16[(size_t)row * H4 + c]          = (_Float16)sr;
    inp16[(size_t)row * H4 + 2 * HH + c] = (_Float16)ctx;
  }
}

// logits = relu(Hout) @ W2^T + b2
__global__ void final_logits_kernel(const float* __restrict__ Hout,
                                    const float* __restrict__ W2, const float* __restrict__ b2,
                                    float* __restrict__ logits) {
  const int row  = blockIdx.x * 8 + (threadIdx.x >> 5);
  const int lane = threadIdx.x & 31;
  float acc = 0.0f;
  for (int i = lane; i < DD; i += 32)
    acc += fmaxf(Hout[(size_t)row * DD + i], 0.0f) * W2[i];
  for (int off = 16; off > 0; off >>= 1) acc += __shfl_xor(acc, off, 32);
  if (lane == 0) logits[row] = acc + b2[0];
}

// ---------------------------------------------------------------------------
// Host launch
// ---------------------------------------------------------------------------
extern "C" void kernel_launch(void* const* d_in, const int* in_sizes, int n_in,
                              void* d_out, int out_size, void* d_ws, size_t ws_size,
                              hipStream_t stream) {
  const int*   word_ids   = (const int*)d_in[0];
  const int*   starts     = (const int*)d_in[1];
  const int*   ends       = (const int*)d_in[2];
  const int*   sent2topic = (const int*)d_in[3];
  const float* embed      = (const float*)d_in[4];
  const float* Wi_f = (const float*)d_in[5];
  const float* Wh_f = (const float*)d_in[6];
  const float* bi_f = (const float*)d_in[7];
  const float* bh_f = (const float*)d_in[8];
  const float* Wi_b = (const float*)d_in[9];
  const float* Wh_b = (const float*)d_in[10];
  const float* bi_b = (const float*)d_in[11];
  const float* bh_b = (const float*)d_in[12];
  const float* W_att = (const float*)d_in[13];
  const float* v_att = (const float*)d_in[14];
  const float* W1    = (const float*)d_in[15];
  const float* b1    = (const float*)d_in[16];
  const float* W2    = (const float*)d_in[17];
  const float* b2    = (const float*)d_in[18];
  float* logits = (float*)d_out;

  char* ws = (char*)d_ws;
  size_t off = 0;
  auto alloc = [&](size_t bytes) -> char* {
    char* p = ws + off;
    off += (bytes + 255) & ~(size_t)255;
    return p;
  };

  _Float16* sent16  = (_Float16*)alloc((size_t)ROWS * EP * 2);
  _Float16* Wi16f   = (_Float16*)alloc((size_t)H3 * EP * 2);
  _Float16* Wi16b   = (_Float16*)alloc((size_t)H3 * EP * 2);
  _Float16* Wh16f   = (_Float16*)alloc((size_t)H3 * HH * 2);
  _Float16* Wh16b   = (_Float16*)alloc((size_t)H3 * HH * 2);
  _Float16* WattT16 = (_Float16*)alloc((size_t)H4 * H4 * 2);
  _Float16* W116    = (_Float16*)alloc((size_t)DD * H4 * 2);
  float* GIf    = (float*)alloc((size_t)ROWS * H3 * 4);
  float* GIb    = (float*)alloc((size_t)ROWS * H3 * 4);
  float* outf   = (float*)alloc((size_t)ROWS * HH * 4);
  float* outb   = (float*)alloc((size_t)ROWS * HH * 4);
  float* topicM = (float*)alloc((size_t)BB * TT * 2 * HH * 4);
  _Float16* catds = (_Float16*)alloc((size_t)ROWS * H4 * 2);
  _Float16* catts = (_Float16*)alloc((size_t)ROWS * H4 * 2);
  float* Yds = (float*)alloc((size_t)ROWS * H4 * 4);
  float* Yts = (float*)alloc((size_t)ROWS * H4 * 4);
  _Float16* inp16 = (_Float16*)alloc((size_t)ROWS * H4 * 2);
  float* Hout = (float*)alloc((size_t)ROWS * DD * 4);

  // --- prep: embeddings + weight conversion/transposition ---
  embed_mean_kernel<<<ROWS, 320, 0, stream>>>(word_ids, embed, sent16);
  convert_pad_f16<<<H3, 256, 0, stream>>>(Wi_f, Wi16f, EE, EP);
  convert_pad_f16<<<H3, 256, 0, stream>>>(Wi_b, Wi16b, EE, EP);
  convert_pad_f16<<<H3, 128, 0, stream>>>(Wh_f, Wh16f, HH, HH);
  convert_pad_f16<<<H3, 128, 0, stream>>>(Wh_b, Wh16b, HH, HH);
  convert_pad_f16<<<DD, 256, 0, stream>>>(W1, W116, H4, H4);
  transpose_to_f16<<<H4, 256, 0, stream>>>(W_att, WattT16, H4, H4);

  // --- GI = sent @ Wi^T + bi  (M=16384, N=384, K=320) ---
  {
    const int waves = (ROWS / 16) * (H3 / 64);
    wmma_gemm_f16<<<waves / 8, 256, 0, stream>>>(sent16, Wi16f, bi_f, GIf,
                                                 ROWS, H3, EP, EP, EP, H3);
    wmma_gemm_f16<<<waves / 8, 256, 0, stream>>>(sent16, Wi16b, bi_b, GIb,
                                                 ROWS, H3, EP, EP, EP, H3);
  }

  // --- bidirectional GRU recurrence (persistent, one block per direction) ---
  (void)hipFuncSetAttribute((const void*)gru_kernel,
                            hipFuncAttributeMaxDynamicSharedMemorySize,
                            GRU_LDS_BYTES);
  gru_kernel<<<2, 1024, GRU_LDS_BYTES, stream>>>(
      GIf, GIb, Wh16f, Wh16b, bh_f, bh_b, outf, outb);

  // --- topic boundary diffs ---
  topic_kernel<<<BB * TT, 256, 0, stream>>>(starts, ends, outf, outb, topicM);

  // --- attention inputs + GEMMs (M=16384, N=512, K=512) ---
  build_cat_kernel<<<ROWS, 256, 0, stream>>>(outf, outb, topicM, sent2topic,
                                             catds, catts);
  {
    const int waves = (ROWS / 16) * (H4 / 64);
    wmma_gemm_f16<<<waves / 8, 256, 0, stream>>>(catds, WattT16, nullptr, Yds,
                                                 ROWS, H4, H4, H4, H4, H4);
    wmma_gemm_f16<<<waves / 8, 256, 0, stream>>>(catts, WattT16, nullptr, Yts,
                                                 ROWS, H4, H4, H4, H4, H4);
  }

  // --- scores, weights, context, decoder input ---
  score_context_kernel<<<ROWS / 8, 256, 0, stream>>>(Yds, Yts, v_att, outf, outb,
                                                     topicM, sent2topic, inp16);

  // --- MLP: Hout = inp @ W1^T + b1 (M=16384, N=128, K=512) ---
  {
    const int waves = (ROWS / 16) * (DD / 64);
    wmma_gemm_f16<<<waves / 8, 256, 0, stream>>>(inp16, W116, b1, Hout,
                                                 ROWS, DD, H4, H4, H4, DD);
  }

  // --- logits ---
  final_logits_kernel<<<ROWS / 8, 256, 0, stream>>>(Hout, W2, b2, logits);
}
